// InterpretableMultiHeadAttention_4492535792230
// MI455X (gfx1250) — compile-verified, matrix operand order checked by reading, not run on hardware
//
#include <hip/hip_runtime.h>

// Problem constants (from reference)
#define B_  4
#define S_  2048
#define D_  256
#define H_  4
#define DK_ 64
#define RS_ (S_ + 4)   // LDS row stride in floats: bank-skew pad (+4 dwords)
                       // row r -> bank group (4*r) mod 64: 16 rows cover all
                       // 64 banks -> conflict-free score stores, ~2-way max
                       // on PV-phase ds_load_b128 A-fragments.

typedef __attribute__((ext_vector_type(16))) _Float16 v16h;
typedef __attribute__((ext_vector_type(8)))  _Float16 v8h;
typedef __attribute__((ext_vector_type(8)))  float    v8f;
typedef __attribute__((ext_vector_type(4)))  float    v4f;

union HF16 { v16h v; v8h h[2]; };

// Build a 16-half (8 VGPR) WMMA fragment from a row (A) or column (B) base
// pointer. Per CDNA5 ISA 16-bit 16x32 layout:
//   lanes 0-15 : halves {0..7, 16..23}
//   lanes 16-31: halves {8..15, 24..31}
// Caller passes p = rowbase + hsel*8; both loads are 16B aligned -> *_load_b128.
static __device__ __forceinline__ v16h frag16(const _Float16* p) {
    HF16 u;
    u.h[0] = *(const v8h*)(p);
    u.h[1] = *(const v8h*)(p + 16);
    return u.v;
}

static __device__ __forceinline__ v8f wmma_f16(v16h a, v16h b, v8f c) {
    // (neg_a, A, neg_b, B, c_mod, C, reuse_a, reuse_b)
    return __builtin_amdgcn_wmma_f32_16x16x32_f16(false, a, false, b,
                                                  (short)0, c, false, false);
}

// ---------------------------------------------------------------------------
// Kernel 1: per-head projections.  One block per (b,s) token row, 256 threads.
//   qsh[b,h,s,e] = 0.125 * sum_d q[b,s,d]*Wq[h,e,d]   (f16, scale folded in)
//   ksh[b,h,s,e] =          sum_d k[b,s,d]*Wk[h,e,d]  (f16)
//   vsT[b,e,s]   =          sum_d v[b,s,d]*Wv[e,d]    (f16, transposed)
// ---------------------------------------------------------------------------
__global__ __launch_bounds__(256) void proj_kernel(
    const float* __restrict__ q, const float* __restrict__ k,
    const float* __restrict__ v, const float* __restrict__ Wq,
    const float* __restrict__ Wk, const float* __restrict__ Wv,
    _Float16* __restrict__ qsh, _Float16* __restrict__ ksh,
    _Float16* __restrict__ vsT) {
    const int row = blockIdx.x;            // b*S + s
    const int b = row / S_;
    const int s = row % S_;

    __shared__ float qrow[D_], krow[D_], vrow[D_];
    for (int i = threadIdx.x; i < D_; i += blockDim.x) {
        qrow[i] = q[(size_t)row * D_ + i];
        krow[i] = k[(size_t)row * D_ + i];
        vrow[i] = v[(size_t)row * D_ + i];
    }
    __syncthreads();

    const int t = threadIdx.x;             // 0..255 -> (h,e)
    const int h = t >> 6;
    const int e = t & (DK_ - 1);

    const float* wq = Wq + ((size_t)h * DK_ + e) * D_;
    const float* wk = Wk + ((size_t)h * DK_ + e) * D_;
    float aq = 0.f, ak = 0.f;
#pragma unroll 8
    for (int d = 0; d < D_; ++d) {
        aq += qrow[d] * wq[d];
        ak += krow[d] * wk[d];
    }
    const size_t oi = (((size_t)b * H_ + h) * S_ + s) * DK_ + e;
    qsh[oi] = (_Float16)(aq * 0.125f);     // 1/sqrt(d_k) folded into Q
    ksh[oi] = (_Float16)ak;

    if (t < DK_) {
        const float* wv = Wv + (size_t)t * D_;
        float av = 0.f;
#pragma unroll 8
        for (int d = 0; d < D_; ++d) av += vrow[d] * wv[d];
        vsT[((size_t)b * DK_ + t) * S_ + s] = (_Float16)av;  // transposed
    }
}

// ---------------------------------------------------------------------------
// Kernel 2: attention.  One wave32 per 16-query row block; loops all 4 heads
// so the head-mean accumulates in the PV WMMA accumulators (no atomics).
// LDS: 16 rows x (2048+4) f32 scores (131.3 KB) -> softmax -> in-place f16.
// ---------------------------------------------------------------------------
__global__ __launch_bounds__(32) void attn_kernel(
    const _Float16* __restrict__ qsh, const _Float16* __restrict__ ksh,
    const _Float16* __restrict__ vsT, float* __restrict__ attn_out,
    float* __restrict__ pooled) {
    __shared__ float sc[16 * RS_];         // 131.3 KB of the 320 KB WGP LDS

    const int blk  = blockIdx.x;           // b*(S/16) + rowtile
    const int b    = blk / (S_ / 16);
    const int s0   = (blk % (S_ / 16)) * 16;
    const int lane = threadIdx.x;
    const int rsel = lane & 15;            // M (A rows) / N (B cols)
    const int hsel = (lane >> 4) & 1;      // which K-half group this lane owns
    const int hoff = hsel * 8;             // half-element offset into rows

    v8f acc0 = {}, acc1 = {}, acc2 = {}, acc3 = {};   // pooled 16x64 (sum over h)

    for (int h = 0; h < H_; ++h) {
        // --- A fragments: qs rows s0..s0+15, K = d_k split into 2 x 32 ---
        const _Float16* qbase =
            qsh + (((size_t)(b * H_ + h)) * S_ + s0 + rsel) * DK_;
        const v16h a0 = frag16(qbase + hoff);        // e = 0..31
        const v16h a1 = frag16(qbase + 32 + hoff);   // e = 32..63

        // --- scores: 128 key tiles of 16, two WMMAs each (K=64) ---
        const _Float16* kb = ksh + ((size_t)(b * H_ + h)) * S_ * DK_;
        for (int tile = 0; tile < S_ / 16; ++tile) {
            const _Float16* krow = kb + ((size_t)(tile * 16 + rsel)) * DK_;
            // hide L2 latency: prefetch the key tile 8 iterations ahead
            __builtin_prefetch(
                kb + ((size_t)((((tile + 8) & (S_ / 16 - 1)) * 16) + rsel)) * DK_,
                0, 1);
            v8f c = {};
            c = wmma_f16(a0, frag16(krow + hoff), c);
            c = wmma_f16(a1, frag16(krow + 32 + hoff), c);
            // D layout: VGPR i -> M = hoff + i, N = rsel
#pragma unroll
            for (int i = 0; i < 8; ++i)
                sc[(hoff + i) * RS_ + tile * 16 + rsel] = c[i];
        }
        __syncthreads();

        // --- fp32 softmax per row + stream attn to HBM + pack f16 in LDS ---
        for (int r = 0; r < 16; ++r) {
            float* row = sc + r * RS_;
            float mx = -3.4e38f;
            for (int j = lane; j < S_; j += 32) mx = fmaxf(mx, row[j]);
#pragma unroll
            for (int off = 16; off > 0; off >>= 1)
                mx = fmaxf(mx, __shfl_xor(mx, off, 32));
            float sum = 0.f;
            for (int j = lane; j < S_; j += 32) {
                float e = __expf(row[j] - mx);
                row[j] = e;
                sum += e;
            }
#pragma unroll
            for (int off = 16; off > 0; off >>= 1)
                sum += __shfl_xor(sum, off, 32);
            const float inv = 1.0f / sum;

            // attn_out[b, s0+r, h, :] — b128 non-temporal streaming stores
            float* aout =
                attn_out + (((size_t)b * S_ + s0 + r) * H_ + h) * S_;
            _Float16* rowh = (_Float16*)row;   // in-place front-pack to f16
            for (int j = lane; j < S_ / 4; j += 32) {
                const float e0 = row[4 * j + 0] * inv;
                const float e1 = row[4 * j + 1] * inv;
                const float e2 = row[4 * j + 2] * inv;
                const float e3 = row[4 * j + 3] * inv;
                v4f st; st[0] = e0; st[1] = e1; st[2] = e2; st[3] = e3;
                __builtin_nontemporal_store(st, (v4f*)(aout + 4 * j));
                // write dwords [2j,2j+1] < read dwords [4j..4j+3]: safe in the
                // in-order single wave (loads of an iteration precede stores).
                rowh[4 * j + 0] = (_Float16)e0;
                rowh[4 * j + 1] = (_Float16)e1;
                rowh[4 * j + 2] = (_Float16)e2;
                rowh[4 * j + 3] = (_Float16)e3;
            }
        }
        __syncthreads();

        // --- PV: attn_f16[16 x 2048] @ vsT -> accumulate heads into acc* ---
        const _Float16* vb = vsT + (size_t)b * DK_ * S_;
        const _Float16* arow = (const _Float16*)(sc + rsel * RS_);
        for (int tile = 0; tile < S_ / 32; ++tile) {
            const v16h pa = frag16(arow + tile * 32 + hoff);
            acc0 = wmma_f16(pa, frag16(vb + ((size_t)( 0 + rsel)) * S_ + tile * 32 + hoff), acc0);
            acc1 = wmma_f16(pa, frag16(vb + ((size_t)(16 + rsel)) * S_ + tile * 32 + hoff), acc1);
            acc2 = wmma_f16(pa, frag16(vb + ((size_t)(32 + rsel)) * S_ + tile * 32 + hoff), acc2);
            acc3 = wmma_f16(pa, frag16(vb + ((size_t)(48 + rsel)) * S_ + tile * 32 + hoff), acc3);
        }
        __syncthreads();
    }

    // pooled[b, s0+m, e] = 0.25 * sum_h heads ;  D layout: M=hoff+i, N=rsel
    float* pb = pooled + ((size_t)b * S_ + s0) * DK_;
#pragma unroll
    for (int i = 0; i < 8; ++i) {
        const int m = hoff + i;
        pb[(size_t)m * DK_ +  0 + rsel] = acc0[i] * 0.25f;
        pb[(size_t)m * DK_ + 16 + rsel] = acc1[i] * 0.25f;
        pb[(size_t)m * DK_ + 32 + rsel] = acc2[i] * 0.25f;
        pb[(size_t)m * DK_ + 48 + rsel] = acc3[i] * 0.25f;
    }
}

// ---------------------------------------------------------------------------
// Kernel 3: out[b,s,f] = sum_e pooled[b,s,e] * Wh[f,e].  256 threads / row.
// ---------------------------------------------------------------------------
__global__ __launch_bounds__(256) void out_kernel(
    const float* __restrict__ pooled, const float* __restrict__ Wh,
    float* __restrict__ out) {
    const int row = blockIdx.x;            // b*S + s
    __shared__ float p[DK_];
    if (threadIdx.x < DK_) p[threadIdx.x] = pooled[(size_t)row * DK_ + threadIdx.x];
    __syncthreads();
    const int f = threadIdx.x;
    const float* w = Wh + (size_t)f * DK_;
    float a = 0.f;
#pragma unroll
    for (int d = 0; d < DK_; ++d) a += p[d] * w[d];
    out[(size_t)row * D_ + f] = a;
}

// ---------------------------------------------------------------------------
extern "C" void kernel_launch(void* const* d_in, const int* in_sizes, int n_in,
                              void* d_out, int out_size, void* d_ws, size_t ws_size,
                              hipStream_t stream) {
    (void)in_sizes; (void)n_in; (void)out_size; (void)ws_size;

    const float* q  = (const float*)d_in[0];
    const float* k  = (const float*)d_in[1];
    const float* v  = (const float*)d_in[2];
    const float* Wq = (const float*)d_in[3];
    const float* Wk = (const float*)d_in[4];
    const float* Wv = (const float*)d_in[5];
    const float* Wh = (const float*)d_in[6];

    // Workspace layout (11 MB total)
    char* ws = (char*)d_ws;
    const size_t qs_bytes  = (size_t)B_ * H_ * S_ * DK_ * sizeof(_Float16); // 4 MB
    const size_t vsT_bytes = (size_t)B_ * DK_ * S_ * sizeof(_Float16);      // 1 MB
    _Float16* qsh    = (_Float16*)(ws);
    _Float16* ksh    = (_Float16*)(ws + qs_bytes);
    _Float16* vsT    = (_Float16*)(ws + 2 * qs_bytes);
    float*    pooled = (float*)   (ws + 2 * qs_bytes + vsT_bytes);          // 2 MB

    float* out  = (float*)d_out;                       // [B,S,256]
    float* attn = out + (size_t)B_ * S_ * D_;          // [B,S,H,S]

    proj_kernel<<<dim3(B_ * S_), dim3(256), 0, stream>>>(q, k, v, Wq, Wk, Wv,
                                                         qsh, ksh, vsT);
    attn_kernel<<<dim3(B_ * (S_ / 16)), dim3(32), 0, stream>>>(qsh, ksh, vsT,
                                                               attn, pooled);
    out_kernel<<<dim3(B_ * S_), dim3(256), 0, stream>>>(pooled, Wh, out);
}
